// EncoderBlock_76493367542449
// MI455X (gfx1250) — matrix operand order repaired and checked by reading, not compiled, not run on hardware
//
#include <hip/hip_runtime.h>
#include <hip/hip_bf16.h>
#include <hip/hip_fp16.h>

typedef __attribute__((ext_vector_type(16))) _Float16 v16h;
typedef __attribute__((ext_vector_type(8)))  float    v8f;

#define DMODEL 1024
#define NH     16
#define DK     64
#define DFF    4096
#define SEQ    1024
#define BATCH  8
#define NTOK   (BATCH * SEQ)
#define EPSF   1e-6f

union FragH {
    v16h v;
    unsigned u[8];
    _Float16 h[16];
};

// ---- WMMA fragment loaders (layouts per CDNA5 ISA 7.12.2, 16-bit, wave32) ----
// A matrix 16x32 (MxK), row-major source with leading dim ld (elements).
// Lanes 0-15: M=lane, V0..3 -> K=0..7, V4..7 -> K=16..23
// Lanes 16-31: M=lane-16, V0..3 -> K=8..15, V4..7 -> K=24..31
__device__ inline void load_frag_a(FragH& f, const _Float16* A, int ld, int k0, int lane) {
    const unsigned* p = reinterpret_cast<const unsigned*>(A + (size_t)(lane & 15) * ld);
    int hi  = lane >> 4;
    int kb1 = (k0 + (hi ? 8 : 0)) >> 1;
    int kb2 = (k0 + (hi ? 24 : 16)) >> 1;
#pragma unroll
    for (int i = 0; i < 4; ++i) f.u[i] = p[kb1 + i];
#pragma unroll
    for (int i = 0; i < 4; ++i) f.u[4 + i] = p[kb2 + i];
}

// B matrix 32x16 (KxN) where B[k][n] = W[n][k] with W row-major [N][K].
// Lanes 0-15: N=lane, V0..3 -> K=0..7, V4..7 -> K=16..23 ; Lanes 16-31: N=lane-16, V0..3 -> K=8..15, V4..7 -> K=24..31
__device__ inline void load_frag_b(FragH& f, const _Float16* W, int ld, int k0, int lane) {
    const unsigned* p = reinterpret_cast<const unsigned*>(W + (size_t)(lane & 15) * ld);
    int kb1 = (k0 + ((lane >> 4) ? 8 : 0)) >> 1;
    int kb2 = (k0 + ((lane >> 4) ? 24 : 16)) >> 1;
#pragma unroll
    for (int i = 0; i < 4; ++i) f.u[i] = p[kb1 + i];
#pragma unroll
    for (int i = 0; i < 4; ++i) f.u[4 + i] = p[kb2 + i];
}

__device__ inline v8f wmma_f16(const FragH& a, const FragH& b, v8f c) {
    return __builtin_amdgcn_wmma_f32_16x16x32_f16(false, a.v, false, b.v,
                                                  (short)0, c, false, false);
}

// ---- fp32 -> fp16 convert ----
__global__ void cvt_f16_kernel(const float* __restrict__ in, _Float16* __restrict__ out, int n) {
    int i = blockIdx.x * blockDim.x + threadIdx.x;
    if (i < n) out[i] = (_Float16)in[i];
}

// ---- LayerNorm (torch-style: ddof=1 std, eps added to std), scalar alpha/beta ----
__global__ __launch_bounds__(256) void ln_kernel(const float* __restrict__ x,
                                                 const float* __restrict__ alpha,
                                                 const float* __restrict__ beta,
                                                 _Float16* __restrict__ out) {
    __shared__ float sred[256];
    int t = blockIdx.x;
    int tid = threadIdx.x;
    const float* xr = x + (size_t)t * DMODEL;

    float s = 0.f;
    for (int i = tid; i < DMODEL; i += 256) s += xr[i];
    sred[tid] = s; __syncthreads();
    for (int o = 128; o > 0; o >>= 1) { if (tid < o) sred[tid] += sred[tid + o]; __syncthreads(); }
    float mean = sred[0] / (float)DMODEL;
    __syncthreads();

    float s2 = 0.f;
    for (int i = tid; i < DMODEL; i += 256) { float d = xr[i] - mean; s2 += d * d; }
    sred[tid] = s2; __syncthreads();
    for (int o = 128; o > 0; o >>= 1) { if (tid < o) sred[tid] += sred[tid + o]; __syncthreads(); }
    float var = sred[0] / (float)(DMODEL - 1);

    float inv = 1.f / (sqrtf(var) + EPSF);
    float a = alpha[0], bb = beta[0];
    for (int i = tid; i < DMODEL; i += 256)
        out[(size_t)t * DMODEL + i] = (_Float16)((xr[i] - mean) * inv * a + bb);
}

// ---- generic GEMM: C[M,N] = act(A[M,K] @ W[N,K]^T + bias) (+ res) ----
// block = 128 threads = 4 waves; block tile = 32 rows x 256 cols; wave tile 32x64
// (2 M-subtiles per wave -> each B fragment feeds 2 WMMAs; halves L2 weight traffic)
template<bool RELU, bool RES, bool OUT16>
__global__ __launch_bounds__(128) void gemm_wmma_kernel(
    const _Float16* __restrict__ A, const _Float16* __restrict__ W,
    const float* __restrict__ bias, const float* __restrict__ res,
    float* __restrict__ out32, _Float16* __restrict__ out16,
    int M, int N, int K) {
    int lane = threadIdx.x & 31;
    int wave = threadIdx.x >> 5;
    int m0 = blockIdx.x * 32;
    int nbase = blockIdx.y * 256 + wave * 64;
    const _Float16* Arow0 = A + (size_t)m0 * K;
    const _Float16* Arow1 = A + (size_t)(m0 + 16) * K;

    v8f acc[2][4];
#pragma unroll
    for (int t = 0; t < 2; ++t)
#pragma unroll
        for (int j = 0; j < 4; ++j)
            acc[t][j] = (v8f){0.f,0.f,0.f,0.f,0.f,0.f,0.f,0.f};

    for (int k0 = 0; k0 < K; k0 += 32) {
        FragH fa0, fa1;
        load_frag_a(fa0, Arow0, K, k0, lane);
        load_frag_a(fa1, Arow1, K, k0, lane);
#pragma unroll
        for (int j = 0; j < 4; ++j) {
            FragH fb; load_frag_b(fb, W + (size_t)(nbase + j * 16) * K, K, k0, lane);
            acc[0][j] = wmma_f16(fa0, fb, acc[0][j]);
            acc[1][j] = wmma_f16(fa1, fb, acc[1][j]);
        }
    }

    int nlane = lane & 15;
    int mo = (lane >> 4) * 8;
#pragma unroll
    for (int t = 0; t < 2; ++t) {
#pragma unroll
        for (int j = 0; j < 4; ++j) {
            int n = nbase + j * 16 + nlane;
            float bv = bias[n];
#pragma unroll
            for (int r = 0; r < 8; ++r) {
                int m = m0 + t * 16 + mo + r;
                float val = acc[t][j][r] + bv;
                if (RELU) val = val > 0.f ? val : 0.f;
                if (RES)  val += res[(size_t)m * N + n];
                if (OUT16) out16[(size_t)m * N + n] = (_Float16)val;
                else       out32[(size_t)m * N + n] = val;
            }
        }
    }
}

// ---- q[t, h*64+d] -> qT[b][h][d][s], LDS-tiled (coalesced read AND write) ----
__global__ __launch_bounds__(256) void qtrans_kernel(const _Float16* __restrict__ q,
                                                     _Float16* __restrict__ qT) {
    __shared__ _Float16 tile[64][65];   // pad to avoid bank conflicts
    int s0 = blockIdx.x * 64;
    int h  = blockIdx.y;
    int b  = blockIdx.z;
    int tid = threadIdx.x;
#pragma unroll
    for (int i = 0; i < 16; ++i) {
        int idx = tid + i * 256;        // 0..4095
        int sl = idx >> 6;              // seq within tile
        int d  = idx & 63;              // head dim (contiguous in global)
        tile[d][sl] = q[((size_t)(b * SEQ + s0 + sl)) * DMODEL + h * DK + d];
    }
    __syncthreads();
#pragma unroll
    for (int i = 0; i < 16; ++i) {
        int idx = tid + i * 256;
        int dl = idx >> 6;
        int sl = idx & 63;              // contiguous in qT
        qT[((size_t)((b * NH + h) * DK + dl)) * SEQ + s0 + sl] = tile[dl][sl];
    }
}

// ---- fused attention: per (b, h, 16-row block): scores -> softmax -> ctx ----
// k == v == q (reference bug reproduced)
__global__ __launch_bounds__(256) void attn_kernel(
    const _Float16* __restrict__ q, const _Float16* __restrict__ qT,
    const int* __restrict__ mask, _Float16* __restrict__ ctx) {
    __shared__ float    s_sc[16 * SEQ];   // 64 KB fp32 scores
    __shared__ _Float16 s_p[16 * SEQ];    // 32 KB fp16 probs
    __shared__ float    s_red[16 * 16];
    __shared__ float    s_stat[16];

    int lane = threadIdx.x & 31;
    int wave = threadIdx.x >> 5;
    int m0 = blockIdx.x * 16;
    int h  = blockIdx.y;
    int b  = blockIdx.z;

    const _Float16* qbase = q + ((size_t)b * SEQ) * DMODEL + h * DK;
    FragH fa0, fa1;
    load_frag_a(fa0, qbase + (size_t)m0 * DMODEL, DMODEL, 0,  lane);
    load_frag_a(fa1, qbase + (size_t)m0 * DMODEL, DMODEL, 32, lane);

    const int* maskb = mask + b * SEQ;
    int nlane = lane & 15;
    int mo = (lane >> 4) * 8;

    // ---- scores = (q @ q^T) * 1/sqrt(dk), masked ----
    for (int jt = wave; jt < SEQ / 16; jt += 8) {
        int j0 = jt * 16;
        FragH fb0, fb1;
        load_frag_b(fb0, qbase + (size_t)j0 * DMODEL, DMODEL, 0,  lane);
        load_frag_b(fb1, qbase + (size_t)j0 * DMODEL, DMODEL, 32, lane);
        v8f acc = (v8f){0.f,0.f,0.f,0.f,0.f,0.f,0.f,0.f};
        acc = wmma_f16(fa0, fb0, acc);
        acc = wmma_f16(fa1, fb1, acc);
        int n = j0 + nlane;
        int mv = maskb[n];
#pragma unroll
        for (int r = 0; r < 8; ++r) {
            float v = acc[r] * 0.125f;   // 1/sqrt(64)
            if (mv == 0) v = -1e9f;
            s_sc[(mo + r) * SEQ + n] = v;
        }
    }
    __syncthreads();

    // ---- softmax over each of the 16 rows (16 threads per row) ----
    int row = threadIdx.x >> 4;
    int sub = threadIdx.x & 15;
    float lmax = -3.4e38f;
    for (int c = sub; c < SEQ; c += 16) lmax = fmaxf(lmax, s_sc[row * SEQ + c]);
    s_red[row * 16 + sub] = lmax;
    __syncthreads();
    if (sub == 0) {
        float m = s_red[row * 16];
        for (int i = 1; i < 16; ++i) m = fmaxf(m, s_red[row * 16 + i]);
        s_stat[row] = m;
    }
    __syncthreads();
    float rmax = s_stat[row];
    float lsum = 0.f;
    for (int c = sub; c < SEQ; c += 16) {
        float e = __expf(s_sc[row * SEQ + c] - rmax);
        s_sc[row * SEQ + c] = e;
        lsum += e;
    }
    s_red[row * 16 + sub] = lsum;
    __syncthreads();
    if (sub == 0) {
        float t = 0.f;
        for (int i = 0; i < 16; ++i) t += s_red[row * 16 + i];
        s_stat[row] = 1.f / t;
    }
    __syncthreads();
    float rinv = s_stat[row];
    for (int c = sub; c < SEQ; c += 16)
        s_p[row * SEQ + c] = (_Float16)(s_sc[row * SEQ + c] * rinv);
    __syncthreads();

    // ---- ctx = P @ q  (B operand from transposed qT; waves 0..3, one 16-col d-tile each)
    if (wave < 4) {
        int n0 = wave * 16;
        const _Float16* qTb = qT + ((size_t)(b * NH + h) * DK) * SEQ;
        v8f acc = (v8f){0.f,0.f,0.f,0.f,0.f,0.f,0.f,0.f};
        for (int j0 = 0; j0 < SEQ; j0 += 32) {
            FragH fa; load_frag_a(fa, s_p, SEQ, j0, lane);
            FragH fb; load_frag_b(fb, qTb + (size_t)n0 * SEQ, SEQ, j0, lane);
            acc = wmma_f16(fa, fb, acc);
        }
        int n = n0 + nlane;
#pragma unroll
        for (int r = 0; r < 8; ++r) {
            int m = m0 + mo + r;
            ctx[((size_t)(b * SEQ + m)) * DMODEL + h * DK + n] = (_Float16)acc[r];
        }
    }
}

extern "C" void kernel_launch(void* const* d_in, const int* in_sizes, int n_in,
                              void* d_out, int out_size, void* d_ws, size_t ws_size,
                              hipStream_t stream) {
    // inputs (setup_inputs order)
    const float* x     = (const float*)d_in[0];
    const int*   mask  = (const int*)  d_in[1];
    const float* wq    = (const float*)d_in[2];
    const float* bq    = (const float*)d_in[3];
    // d_in[4..7]: wk,bk,wv,bv -> outputs discarded in reference; skipped
    const float* wo    = (const float*)d_in[8];
    const float* bo    = (const float*)d_in[9];
    const float* w1    = (const float*)d_in[10];
    const float* b1    = (const float*)d_in[11];
    const float* w2    = (const float*)d_in[12];
    const float* b2    = (const float*)d_in[13];
    const float* ln1a  = (const float*)d_in[14];
    const float* ln1b  = (const float*)d_in[15];
    const float* ln2a  = (const float*)d_in[16];
    const float* ln2b  = (const float*)d_in[17];
    float* out = (float*)d_out;

    // workspace layout (fp16 buffers)
    char* ws = (char*)d_ws;
    const size_t MB = 1024ull * 1024ull;
    _Float16* wq16  = (_Float16*)(ws + 0 * MB);    // 2 MB
    _Float16* wo16  = (_Float16*)(ws + 2 * MB);    // 2 MB
    _Float16* w116  = (_Float16*)(ws + 4 * MB);    // 8 MB
    _Float16* w216  = (_Float16*)(ws + 12 * MB);   // 8 MB
    _Float16* n16   = (_Float16*)(ws + 20 * MB);   // 16 MB (reused for LN2)
    _Float16* q16   = (_Float16*)(ws + 36 * MB);   // 16 MB
    _Float16* qT16  = (_Float16*)(ws + 52 * MB);   // 16 MB
    _Float16* ctx16 = (_Float16*)(ws + 68 * MB);   // 16 MB
    _Float16* h16   = (_Float16*)(ws + 84 * MB);   // 64 MB

    // 1) weights fp32 -> fp16 (skip wk/wv: dead code in reference)
    cvt_f16_kernel<<<(DMODEL * DMODEL + 255) / 256, 256, 0, stream>>>(wq, wq16, DMODEL * DMODEL);
    cvt_f16_kernel<<<(DMODEL * DMODEL + 255) / 256, 256, 0, stream>>>(wo, wo16, DMODEL * DMODEL);
    cvt_f16_kernel<<<(DFF * DMODEL + 255) / 256, 256, 0, stream>>>(w1, w116, DFF * DMODEL);
    cvt_f16_kernel<<<(DMODEL * DFF + 255) / 256, 256, 0, stream>>>(w2, w216, DMODEL * DFF);

    // 2) n = LN1(x)
    ln_kernel<<<NTOK, 256, 0, stream>>>(x, ln1a, ln1b, n16);

    // 3) q = n @ wq^T + bq  (fp16 out)
    gemm_wmma_kernel<false, false, true><<<dim3(NTOK / 32, DMODEL / 256), 128, 0, stream>>>(
        n16, wq16, bq, nullptr, nullptr, q16, NTOK, DMODEL, DMODEL);

    // 4) qT[b][h][d][s]
    qtrans_kernel<<<dim3(SEQ / 64, NH, BATCH), 256, 0, stream>>>(q16, qT16);

    // 5) attention (k=v=q), fused scores/softmax/ctx
    attn_kernel<<<dim3(SEQ / 16, NH, BATCH), 256, 0, stream>>>(q16, qT16, mask, ctx16);

    // 6) out = x + ctx @ wo^T + bo   (fp32, residual = x)
    gemm_wmma_kernel<false, true, false><<<dim3(NTOK / 32, DMODEL / 256), 128, 0, stream>>>(
        ctx16, wo16, bo, x, out, nullptr, NTOK, DMODEL, DMODEL);

    // 7) n2 = LN2(out)
    ln_kernel<<<NTOK, 256, 0, stream>>>(out, ln2a, ln2b, n16);

    // 8) h = relu(n2 @ w1^T + b1)    (fp16 out)
    gemm_wmma_kernel<true, false, true><<<dim3(NTOK / 32, DFF / 256), 128, 0, stream>>>(
        n16, w116, b1, nullptr, nullptr, h16, NTOK, DFF, DMODEL);

    // 9) out = out + h @ w2^T + b2   (fp32, in-place residual)
    gemm_wmma_kernel<false, true, false><<<dim3(NTOK / 32, DMODEL / 256), 128, 0, stream>>>(
        h16, w216, b2, out, out, nullptr, NTOK, DMODEL, DFF);
}
